// SurfaceTransolver_12481174962457
// MI455X (gfx1250) — matrix operand order repaired and checked by reading, not compiled, not run on hardware
//
#include <hip/hip_runtime.h>
#include <hip/hip_bf16.h>

// ---------------------------------------------------------------------------
// SurfaceTransolver on MI455X (gfx1250): bf16 WMMA pipeline, fp32 accumulate,
// TDM (tensor_load_to_lds) weight staging, b128-vectorized LDS fragments.
// Dims: B=1, N=200000, HID=256, NH=8, S=64, DH=32.
// ---------------------------------------------------------------------------

#define NTOK 200000
#define HIDD 256
#define NHEAD 8
#define NSLICE 64
#define DHEAD 32

typedef __attribute__((ext_vector_type(16))) __bf16 v16bf;
typedef __attribute__((ext_vector_type(8)))  __bf16 v8bf;
typedef __attribute__((ext_vector_type(4)))  __bf16 v4bf;
typedef __attribute__((ext_vector_type(8)))  float  v8f;
typedef __attribute__((ext_vector_type(4)))  unsigned int u32x4;
typedef __attribute__((ext_vector_type(4)))  int i32x4;
typedef __attribute__((ext_vector_type(8)))  int i32x8;

#if __has_builtin(__builtin_amdgcn_tensor_load_to_lds)
#define USE_TDM 1
#else
#define USE_TDM 0
#endif

__device__ __forceinline__ v8f wmma_bf16(v16bf a, v16bf b, v8f c) {
  return __builtin_amdgcn_wmma_f32_16x16x32_bf16(false, a, false, b, (short)0, c,
                                                 false, false);
}

// A fragment (16x32 MxK) from row-major p[m*ld + k].
// ISA layout -> per lane: two contiguous 8-elem chunks at k = lh*8 and 16+lh*8.
__device__ __forceinline__ v16bf load_a(const __bf16* p, int ld) {
  const int l = threadIdx.x & 31;
  const int m = l & 15, lh = l >> 4;
  const v8bf lo = *(const v8bf*)(p + m * ld + lh * 8);
  const v8bf hi = *(const v8bf*)(p + m * ld + 16 + lh * 8);
  v16bf a;
#pragma unroll
  for (int i = 0; i < 8; ++i) { a[i] = lo[i]; a[8 + i] = hi[i]; }
  return a;
}

// B fragment (32x16 KxN) from an N-major (transposed) source: B[k][n] = p[n*ld + k].
// Per lane: two contiguous 8-elem chunks at k = lh*8 and 16+lh*8 -> two 16B loads.
__device__ __forceinline__ v16bf load_bt(const __bf16* p, int ld) {
  const int l = threadIdx.x & 31;
  const int n = l & 15, lh = l >> 4;
  const v8bf lo = *(const v8bf*)(p + n * ld + lh * 8);
  const v8bf hi = *(const v8bf*)(p + n * ld + 16 + lh * 8);
  v16bf b;
#pragma unroll
  for (int i = 0; i < 8; ++i) { b[i] = lo[i]; b[8 + i] = hi[i]; }
  return b;
}

// ---------------------------------------------------------------------------
// Tensor Data Mover: load a 2D bf16 tile (tile_k contiguous elems x tile_n
// rows, row stride = stride elems) into LDS with 16DW+4DW padding (ld 40).
// D# layout per CDNA5 ISA ch.8 (group0 128b, group1 256b; groups 2/3 zero).
// ---------------------------------------------------------------------------
__device__ __forceinline__ void tdm_load_tile_bf16(unsigned lds_addr,
                                                   const __bf16* gaddr,
                                                   int tile_k, int tile_n,
                                                   int stride) {
#if USE_TDM
  const unsigned long long ga = (unsigned long long)(size_t)gaddr;
  u32x4 g0;
  g0[0] = 1u;                                      // count=1, user descriptor
  g0[1] = lds_addr;                                // lds_addr [63:32]
  g0[2] = (unsigned)ga;                            // global_addr [95:64]
  g0[3] = (unsigned)((ga >> 32) & 0x1FFFFFFull)    // global_addr [120:96]
        | (2u << 30);                              // type=2 ("image")
  i32x8 g1;
  g1[0] = (1 << 16)      // data_size = 2 bytes
        | (1 << 20)      // pad_enable
        | (3 << 22)      // pad_interval: 16 DWORDs (= 32 bf16 of data)
        | (3 << 25);     // pad_amount:   4 DWORDs (= 8 bf16 -> ld 40)
  g1[1] = 0;                          // tensor_dim0 = 1<<20 (low16 = 0)
  g1[2] = 0x10;                       // tensor_dim0 hi16; tensor_dim1 low16 = 0
  g1[3] = 0x10 | (tile_k << 16);      // tensor_dim1 hi16; tile_dim0
  g1[4] = tile_n;                     // tile_dim1; tile_dim2 = 0
  g1[5] = stride;                     // tensor_dim0_stride low 32
  g1[6] = 0;
  g1[7] = 0;
  const i32x4 z4 = {0, 0, 0, 0};
  const i32x8 z8 = {0, 0, 0, 0, 0, 0, 0, 0};
  __builtin_amdgcn_tensor_load_to_lds(g0, g1, z4, z4, z8, 0);
  __builtin_amdgcn_s_wait_tensorcnt((short)0);
#endif
}

// ---------------------------------------------------------------------------
// K0: weight prep -> bf16, TRANSPOSED so B tiles are N-major:
//   WcbT[j][k] = [W_fx | W_x][k][j]   (512 x 256)
//   WpbT[j][k] = W_proj[k][j]         (256 x 256)
// ---------------------------------------------------------------------------
__global__ void k_prep_weights(const float* __restrict__ Wfx, const float* __restrict__ Wx,
                               const float* __restrict__ Wproj,
                               __bf16* __restrict__ WcbT, __bf16* __restrict__ WpbT) {
  const int idx = blockIdx.x * blockDim.x + threadIdx.x;
  const int stride = gridDim.x * blockDim.x;
  for (int i = idx; i < 512 * 256; i += stride) {
    const int j = i >> 8, k = i & 255;
    const float v = (j < 256) ? Wfx[k * 256 + j] : Wx[k * 256 + (j - 256)];
    WcbT[i] = (__bf16)v;
  }
  for (int i = idx; i < 256 * 256; i += stride) {
    const int j = i >> 8, k = i & 255;
    WpbT[i] = (__bf16)Wproj[k * 256 + j];
  }
}

__global__ void k_zero(float* __restrict__ num, float* __restrict__ norm) {
  const int idx = blockIdx.x * blockDim.x + threadIdx.x;
  const int stride = gridDim.x * blockDim.x;
  for (int i = idx; i < NHEAD * NSLICE * DHEAD; i += stride) num[i] = 0.f;
  for (int i = idx; i < NHEAD * NSLICE; i += stride) norm[i] = 0.f;
}

// ---------------------------------------------------------------------------
// K1: yb[n][0:512] = x @ [W_fx | W_x] + bias  (bf16 out, WMMA fp32 accum)
// Block: 64 rows x 256 cols; weight tile staged per K-step by the TDM.
// ---------------------------------------------------------------------------
__global__ __launch_bounds__(256) void k_gemm1(const float* __restrict__ x,
                                               const __bf16* __restrict__ WcbT,
                                               const float* __restrict__ bfx,
                                               const float* __restrict__ bx,
                                               __bf16* __restrict__ yb) {
  __shared__ __align__(16) __bf16 As[64][40];    // 64 rows x 32 K
  __shared__ __align__(16) __bf16 BsT[256][40];  // 256 cols x 32 K (N-major)
  const int row0 = blockIdx.x * 64;
  const int col0 = blockIdx.y * 256;
  const int tid = threadIdx.x;
  const int wid = tid >> 5, lane = tid & 31;
  const int rstrip = (wid & 3) * 16;
  const int chalf = (wid >> 2) * 128;

  v8f acc[8];
#pragma unroll
  for (int t = 0; t < 8; ++t)
#pragma unroll
    for (int r = 0; r < 8; ++r) acc[t][r] = 0.f;

  for (int kk = 0; kk < 256; kk += 32) {
    if (kk + 32 < 256)
      __builtin_prefetch(&x[(size_t)row0 * HIDD + kk + 32], 0, 0);
    // A: 64x32 fp32 -> bf16, b128 loads
    for (int i = tid; i < 512; i += 256) {
      const int r = i >> 3, c4 = (i & 7) * 4;
      const float4 f = *(const float4*)&x[(size_t)(row0 + r) * HIDD + kk + c4];
      v4bf o;
      o[0] = (__bf16)f.x; o[1] = (__bf16)f.y; o[2] = (__bf16)f.z; o[3] = (__bf16)f.w;
      *(v4bf*)&As[r][c4] = o;
    }
    // B: 256 cols x 32 K tile of WcbT -> LDS via Tensor Data Mover (wave 0)
#if USE_TDM
    if (wid == 0)
      tdm_load_tile_bf16((unsigned)(size_t)&BsT[0][0],
                         &WcbT[(size_t)col0 * 256 + kk], 32, 256, 256);
#else
    for (int i = tid; i < 1024; i += 256) {
      const int j = i >> 2, c8 = (i & 3) * 8;
      *(v8bf*)&BsT[j][c8] = *(const v8bf*)&WcbT[(size_t)(col0 + j) * 256 + kk + c8];
    }
#endif
    __syncthreads();
    const v16bf a = load_a(&As[rstrip][0], 40);
#pragma unroll
    for (int t = 0; t < 8; ++t) {
      const v16bf b = load_bt(&BsT[chalf + t * 16][0], 40);
      acc[t] = wmma_bf16(a, b, acc[t]);
    }
    __syncthreads();
  }

  const int n = lane & 15, mh = (lane >> 4) * 8;
#pragma unroll
  for (int t = 0; t < 8; ++t) {
#pragma unroll
    for (int r = 0; r < 8; ++r) {
      const int row = row0 + rstrip + mh + r;
      const int j = col0 + chalf + t * 16 + n;
      const float bias = (j < 256) ? bfx[j] : bx[j - 256];
      yb[(size_t)row * 512 + j] = (__bf16)(acc[t][r] + bias);
    }
  }
}

// ---------------------------------------------------------------------------
// K2: per-(token,head) slice softmax with temperature + mask -> swb bf16
// ---------------------------------------------------------------------------
__global__ __launch_bounds__(256) void k_slice_softmax(const __bf16* __restrict__ yb,
                                                       const float* __restrict__ Wslice,
                                                       const float* __restrict__ bslice,
                                                       const float* __restrict__ temp,
                                                       const float* __restrict__ mask,
                                                       __bf16* __restrict__ swb) {
  __shared__ float Ws[32][64];
  __shared__ float bs[64];
  for (int i = threadIdx.x; i < 32 * 64; i += 256) Ws[i >> 6][i & 63] = Wslice[i];
  if (threadIdx.x < 64) bs[threadIdx.x] = bslice[threadIdx.x];
  __syncthreads();

  const int idx = blockIdx.x * 256 + threadIdx.x;
  const int nn = idx >> 3, h = idx & 7;
  if (nn >= NTOK) return;

  float xm[32];
#pragma unroll
  for (int c8 = 0; c8 < 4; ++c8) {
    const v8bf v = *(const v8bf*)&yb[(size_t)nn * 512 + 256 + h * 32 + c8 * 8];
#pragma unroll
    for (int q = 0; q < 8; ++q) xm[c8 * 8 + q] = (float)v[q];
  }

  const float it = 1.0f / temp[h];
  float lg[64];
  float mx = -1e30f;
  for (int s = 0; s < 64; ++s) {
    float a = bs[s];
#pragma unroll
    for (int c = 0; c < 32; ++c) a += xm[c] * Ws[c][s];
    a *= it;
    lg[s] = a;
    mx = fmaxf(mx, a);
  }
  float sum = 0.f;
  for (int s = 0; s < 64; ++s) { lg[s] = __expf(lg[s] - mx); sum += lg[s]; }
  const float inv = mask[nn] / sum;
#pragma unroll
  for (int s8 = 0; s8 < 8; ++s8) {
    v8bf o;
#pragma unroll
    for (int q = 0; q < 8; ++q) o[q] = (__bf16)(lg[s8 * 8 + q] * inv);
    *(v8bf*)&swb[(size_t)nn * 512 + h * 64 + s8 * 8] = o;
  }
}

// ---------------------------------------------------------------------------
// K3: pooling.  num[h][s][c] = sum_n sw[n][h][s]*fx[n][h][c];
//     norm[h][s] = sum_n sw[n][h][s].   K-minor LDS tiles -> b128 fragments.
// Block = 2048 tokens, wave w == head w.
// ---------------------------------------------------------------------------
__global__ __launch_bounds__(256) void k_pool(const __bf16* __restrict__ swb,
                                              const __bf16* __restrict__ yb,
                                              float* __restrict__ num,
                                              float* __restrict__ norm) {
  __shared__ __align__(16) __bf16 SwT[512][40];  // [s-col][token]  (40KB)
  __shared__ __align__(16) __bf16 FxT[256][40];  // [c-col][token]  (20KB)
  const int base = blockIdx.x * 2048;
  const int tid = threadIdx.x, lane = tid & 31;
  const int h = tid >> 5;

  v8f acc[4][2];
#pragma unroll
  for (int st = 0; st < 4; ++st)
#pragma unroll
    for (int ct = 0; ct < 2; ++ct)
#pragma unroll
      for (int r = 0; r < 8; ++r) acc[st][ct][r] = 0.f;
  float nacc0 = 0.f, nacc1 = 0.f;

  for (int step = 0; step < 64; ++step) {
    const int t0 = base + step * 32;
    // transpose-stage slice weights: rows = token, cols = s
    for (int i = tid; i < 2048; i += 256) {
      const int r = i >> 6, c8 = (i & 63) * 8;
      const int tok = t0 + r;
      v8bf v;
      if (tok < NTOK) v = *(const v8bf*)&swb[(size_t)tok * 512 + c8];
      else {
#pragma unroll
        for (int q = 0; q < 8; ++q) v[q] = (__bf16)0.f;
      }
#pragma unroll
      for (int q = 0; q < 8; ++q) SwT[c8 + q][r] = v[q];
    }
    // transpose-stage fx_mid
    for (int i = tid; i < 1024; i += 256) {
      const int r = i >> 5, c8 = (i & 31) * 8;
      const int tok = t0 + r;
      v8bf v;
      if (tok < NTOK) v = *(const v8bf*)&yb[(size_t)tok * 512 + c8];
      else {
#pragma unroll
        for (int q = 0; q < 8; ++q) v[q] = (__bf16)0.f;
      }
#pragma unroll
      for (int q = 0; q < 8; ++q) FxT[c8 + q][r] = v[q];
    }
    __syncthreads();
    const v16bf b0 = load_bt(&FxT[h * 32][0], 40);
    const v16bf b1 = load_bt(&FxT[h * 32 + 16][0], 40);
#pragma unroll
    for (int st = 0; st < 4; ++st) {
      const v16bf a = load_a(&SwT[h * 64 + st * 16][0], 40);  // A = sw^T
      acc[st][0] = wmma_bf16(a, b0, acc[st][0]);
      acc[st][1] = wmma_bf16(a, b1, acc[st][1]);
    }
    // norm partials: row sums of SwT (b128 reads)
#pragma unroll
    for (int c8 = 0; c8 < 32; c8 += 8) {
      const v8bf u0 = *(const v8bf*)&SwT[h * 64 + lane][c8];
      const v8bf u1 = *(const v8bf*)&SwT[h * 64 + 32 + lane][c8];
#pragma unroll
      for (int q = 0; q < 8; ++q) { nacc0 += (float)u0[q]; nacc1 += (float)u1[q]; }
    }
    __syncthreads();
  }

  const int n = lane & 15, mh = (lane >> 4) * 8;
#pragma unroll
  for (int st = 0; st < 4; ++st)
#pragma unroll
    for (int ct = 0; ct < 2; ++ct)
#pragma unroll
      for (int r = 0; r < 8; ++r) {
        const int s = st * 16 + mh + r;
        const int c = ct * 16 + n;
        atomicAdd(&num[(h * NSLICE + s) * DHEAD + c], acc[st][ct][r]);
      }
  atomicAdd(&norm[h * NSLICE + lane], nacc0);
  atomicAdd(&norm[h * NSLICE + 32 + lane], nacc1);
}

// ---------------------------------------------------------------------------
// K4: tiny per-head SDPA; writes out_slice TRANSPOSED: osbT[h][c][s]
// ---------------------------------------------------------------------------
__global__ __launch_bounds__(256) void k_attn(const float* __restrict__ num,
                                              const float* __restrict__ norm,
                                              const float* __restrict__ Wqkv,
                                              __bf16* __restrict__ osbT) {
  __shared__ float st[64][32];
  __shared__ float q[64][32], kk_[64][32], v[64][32];
  __shared__ float attn[64][64];
  const int h = blockIdx.x;
  const int tid = threadIdx.x;

  for (int i = tid; i < 64 * 32; i += 256) {
    const int s = i >> 5, c = i & 31;
    st[s][c] = num[(h * 64 + s) * 32 + c] / (norm[h * 64 + s] + 1e-5f);
  }
  __syncthreads();
  for (int i = tid; i < 64 * 96; i += 256) {
    const int s = i / 96, jc = i % 96;
    float a = 0.f;
#pragma unroll
    for (int c = 0; c < 32; ++c) a += st[s][c] * Wqkv[c * 96 + jc];
    const int part = jc >> 5, cc = jc & 31;
    if (part == 0) q[s][cc] = a;
    else if (part == 1) kk_[s][cc] = a;
    else v[s][cc] = a;
  }
  __syncthreads();
  const float scale = 0.1767766952966369f;  // 1/sqrt(32)
  for (int i = tid; i < 64 * 64; i += 256) {
    const int s = i >> 6, t = i & 63;
    float a = 0.f;
#pragma unroll
    for (int c = 0; c < 32; ++c) a += q[s][c] * kk_[t][c];
    attn[s][t] = a * scale;
  }
  __syncthreads();
  if (tid < 64) {
    float mx = -1e30f;
    for (int t = 0; t < 64; ++t) mx = fmaxf(mx, attn[tid][t]);
    float sum = 0.f;
    for (int t = 0; t < 64; ++t) {
      const float e = __expf(attn[tid][t] - mx);
      attn[tid][t] = e;
      sum += e;
    }
    const float inv = 1.f / sum;
    for (int t = 0; t < 64; ++t) attn[tid][t] *= inv;
  }
  __syncthreads();
  for (int i = tid; i < 64 * 32; i += 256) {
    const int s = i >> 5, c = i & 31;
    float a = 0.f;
#pragma unroll
    for (int t = 0; t < 64; ++t) a += attn[s][t] * v[t][c];
    osbT[(h * 32 + c) * 64 + s] = (__bf16)a;  // transposed: [h][c][s]
  }
}

// ---------------------------------------------------------------------------
// K5: fused scatter + projection.
// Stage 1: outx = per-head sw @ out_slice (A from global, B = osbT), masked
//          -> bf16 LDS tile.
// Stage 2: out = outx @ W_proj + b_proj, * mask (W tile staged by TDM).
// ---------------------------------------------------------------------------
__global__ __launch_bounds__(256) void k_scatter_proj(const __bf16* __restrict__ swb,
                                                      const __bf16* __restrict__ osbT,
                                                      const __bf16* __restrict__ WpbT,
                                                      const float* __restrict__ bproj,
                                                      const float* __restrict__ mask,
                                                      float* __restrict__ out) {
  __shared__ __align__(16) __bf16 Ox[64][264];   // 33.8KB
  __shared__ __align__(16) __bf16 BsT[256][40];  // 20.5KB
  const int row0 = blockIdx.x * 64;
  const int tid = threadIdx.x, lane = tid & 31;
  const int wid = tid >> 5;
  const int rstrip = (wid & 3) * 16;
  const int chalf = (wid >> 2) * 128;
  const int n = lane & 15, mh = (lane >> 4) * 8;

  // ---- Stage 1 ----
#pragma unroll
  for (int ct = 0; ct < 8; ++ct) {
    const int col = chalf + ct * 16;  // 0..255
    const int h = col >> 5;
    const int cc = col & 31;          // 0 or 16
    const v16bf a0 = load_a(&swb[(size_t)(row0 + rstrip) * 512 + h * 64], 512);
    const v16bf a1 = load_a(&swb[(size_t)(row0 + rstrip) * 512 + h * 64 + 32], 512);
    const __bf16* obase = &osbT[(size_t)(h * 32 + cc) * 64];
    const v16bf b0 = load_bt(obase, 64);       // K = 0..31
    const v16bf b1 = load_bt(obase + 32, 64);  // K = 32..63
    v8f acc;
#pragma unroll
    for (int r = 0; r < 8; ++r) acc[r] = 0.f;
    acc = wmma_bf16(a0, b0, acc);
    acc = wmma_bf16(a1, b1, acc);
#pragma unroll
    for (int r = 0; r < 8; ++r) {
      const int row = row0 + rstrip + mh + r;
      Ox[rstrip + mh + r][col + n] = (__bf16)(acc[r] * mask[row]);
    }
  }
  __syncthreads();

  // ---- Stage 2 ----
  v8f acc2[8];
#pragma unroll
  for (int t = 0; t < 8; ++t)
#pragma unroll
    for (int r = 0; r < 8; ++r) acc2[t][r] = 0.f;

  for (int kk = 0; kk < 256; kk += 32) {
#if USE_TDM
    if (wid == 0)
      tdm_load_tile_bf16((unsigned)(size_t)&BsT[0][0], &WpbT[kk], 32, 256, 256);
#else
    for (int i = tid; i < 1024; i += 256) {
      const int j = i >> 2, c8 = (i & 3) * 8;
      *(v8bf*)&BsT[j][c8] = *(const v8bf*)&WpbT[(size_t)j * 256 + kk + c8];
    }
#endif
    __syncthreads();
    const v16bf a = load_a(&Ox[rstrip][kk], 264);
#pragma unroll
    for (int t = 0; t < 8; ++t) {
      const v16bf b = load_bt(&BsT[chalf + t * 16][0], 40);
      acc2[t] = wmma_bf16(a, b, acc2[t]);
    }
    __syncthreads();
  }
#pragma unroll
  for (int t = 0; t < 8; ++t) {
#pragma unroll
    for (int r = 0; r < 8; ++r) {
      const int row = row0 + rstrip + mh + r;
      const int j = chalf + t * 16 + n;
      out[(size_t)row * 256 + j] = (acc2[t][r] + bproj[j]) * mask[row];
    }
  }
}

// ---------------------------------------------------------------------------
// Launch
// ---------------------------------------------------------------------------
extern "C" void kernel_launch(void* const* d_in, const int* in_sizes, int n_in,
                              void* d_out, int out_size, void* d_ws, size_t ws_size,
                              hipStream_t stream) {
  const float* x      = (const float*)d_in[0];
  const float* mask   = (const float*)d_in[1];
  const float* W_fx   = (const float*)d_in[2];
  const float* b_fx   = (const float*)d_in[3];
  const float* W_x    = (const float*)d_in[4];
  const float* b_x    = (const float*)d_in[5];
  const float* W_sl   = (const float*)d_in[6];
  const float* b_sl   = (const float*)d_in[7];
  const float* W_qkv  = (const float*)d_in[8];
  const float* W_proj = (const float*)d_in[9];
  const float* b_proj = (const float*)d_in[10];
  const float* temp   = (const float*)d_in[11];
  float* out = (float*)d_out;

  char* ws = (char*)d_ws;
  const size_t yb_off   = 0;                                   // bf16 [N][512]
  const size_t swb_off  = yb_off  + (size_t)NTOK * 512 * 2;    // bf16 [N][512]
  const size_t wcb_off  = swb_off + (size_t)NTOK * 512 * 2;    // bf16 [512][256] (T)
  const size_t wpb_off  = wcb_off + 512 * 256 * 2;             // bf16 [256][256] (T)
  const size_t num_off  = wpb_off + 256 * 256 * 2;             // f32 [8][64][32]
  const size_t norm_off = num_off + NHEAD * NSLICE * DHEAD * 4;// f32 [8][64]
  const size_t osb_off  = norm_off + NHEAD * NSLICE * 4;       // bf16 [8][32][64] (T)

  __bf16* yb   = (__bf16*)(ws + yb_off);
  __bf16* swb  = (__bf16*)(ws + swb_off);
  __bf16* WcbT = (__bf16*)(ws + wcb_off);
  __bf16* WpbT = (__bf16*)(ws + wpb_off);
  float*  num  = (float*)(ws + num_off);
  float*  norm = (float*)(ws + norm_off);
  __bf16* osbT = (__bf16*)(ws + osb_off);

  k_prep_weights<<<256, 256, 0, stream>>>(W_fx, W_x, W_proj, WcbT, WpbT);
  k_zero<<<64, 256, 0, stream>>>(num, norm);
  k_gemm1<<<dim3(NTOK / 64, 2), 256, 0, stream>>>(x, WcbT, b_fx, b_x, yb);
  k_slice_softmax<<<(NTOK * NHEAD + 255) / 256, 256, 0, stream>>>(yb, W_sl, b_sl, temp,
                                                                  mask, swb);
  k_pool<<<(NTOK + 2047) / 2048, 256, 0, stream>>>(swb, yb, num, norm);
  k_attn<<<NHEAD, 256, 0, stream>>>(num, norm, W_qkv, osbT);
  k_scatter_proj<<<NTOK / 64, 256, 0, stream>>>(swb, osbT, WpbT, b_proj, mask, out);
}
